// MoELayer_22625887716030
// MI455X (gfx1250) — matrix operand order repaired and checked by reading, not compiled, not run on hardware
//
#include <hip/hip_runtime.h>
#include <hip/hip_bf16.h>

// Problem dims (fixed by reference)
#define TOKENS 8192      // B*S
#define DDIM   1024
#define HDIM   2048
#define NEXP   8
#define NPAIR  (TOKENS * 2)   // exactly K=2 experts per token

// GEMM tiling: block = 256 threads = 8 waves; tile 64(M) x 128(N), K-step 32
#define BM 64
#define BN 128
#define BK 32
#define LDK (BK + 8)     // padded LDS row (80B, 16B-aligned rows)

typedef __bf16 bf16;
typedef __attribute__((ext_vector_type(16))) __bf16 v16bf;
typedef __attribute__((ext_vector_type(8)))  float  v8f;

union Frag32B { v16bf v; uint4 q[2]; };
union Pack16B { bf16 h[8]; uint4 q; };
union Pack4B  { bf16 h[2]; unsigned u; };

// ---------------------------------------------------------------- routing ---
__global__ __launch_bounds__(256) void moe_zero_counts(int* counts) {
    if (threadIdx.x < NEXP) counts[threadIdx.x] = 0;
}

__global__ __launch_bounds__(256) void moe_routing(
    const float* __restrict__ x, const float* __restrict__ Wg,
    const float* __restrict__ bg, int* __restrict__ counts,
    int* __restrict__ tok_e, int* __restrict__ tok_rank,
    float* __restrict__ tok_gate)
{
    const int lane = threadIdx.x & 31;
    const int wave = threadIdx.x >> 5;
    const int t = blockIdx.x * 8 + wave;          // one wave32 per token
    if (t >= TOKENS) return;

    float acc[NEXP];
    #pragma unroll
    for (int e = 0; e < NEXP; ++e) acc[e] = 0.0f;

    const float* xr = x + (size_t)t * DDIM;
    for (int d = lane; d < DDIM; d += 32) {
        const float xv = xr[d];
        const float* wg = Wg + (size_t)d * NEXP;
        #pragma unroll
        for (int e = 0; e < NEXP; ++e) acc[e] += xv * wg[e];
    }
    // butterfly reduce across the wave
    #pragma unroll
    for (int off = 16; off > 0; off >>= 1) {
        #pragma unroll
        for (int e = 0; e < NEXP; ++e) acc[e] += __shfl_xor(acc[e], off, 32);
    }

    if (lane == 0) {
        float l[NEXP], mx = -3.4e38f;
        #pragma unroll
        for (int e = 0; e < NEXP; ++e) {
            l[e] = (acc[e] + bg[e]) * 0.2f;       // logits / TEMP, TEMP=5
            mx = fmaxf(mx, l[e]);
        }
        float p[NEXP], s = 0.0f;
        #pragma unroll
        for (int e = 0; e < NEXP; ++e) { p[e] = __expf(l[e] - mx); s += p[e]; }
        const float inv_s = 1.0f / s;
        // top-2 (ties -> lowest index, matching lax.top_k)
        int i0 = 0;
        #pragma unroll
        for (int e = 1; e < NEXP; ++e) if (p[e] > p[i0]) i0 = e;
        int i1 = (i0 == 0) ? 1 : 0;
        #pragma unroll
        for (int e = 0; e < NEXP; ++e) if (e != i0 && p[e] > p[i1]) i1 = e;
        const float v0 = p[i0] * inv_s, v1 = p[i1] * inv_s;
        const float g = 1.0f / (v0 + v1 + 1e-9f);
        const int r0 = atomicAdd(&counts[i0], 1);
        const int r1 = atomicAdd(&counts[i1], 1);
        tok_e[2 * t] = i0;        tok_e[2 * t + 1] = i1;
        tok_rank[2 * t] = r0;     tok_rank[2 * t + 1] = r1;
        tok_gate[2 * t] = v0 * g; tok_gate[2 * t + 1] = v1 * g;
    }
}

__global__ __launch_bounds__(32) void moe_scan(const int* __restrict__ counts,
                                               int* __restrict__ offsets) {
    if (threadIdx.x == 0) {
        int s = 0;
        for (int e = 0; e < NEXP; ++e) { offsets[e] = s; s += counts[e]; }
    }
}

__global__ __launch_bounds__(256) void moe_scatter(
    const int* __restrict__ tok_e, const int* __restrict__ tok_rank,
    const int* __restrict__ offsets, int* __restrict__ slot_token)
{
    const int t = blockIdx.x * 256 + threadIdx.x;
    if (t >= TOKENS) return;
    #pragma unroll
    for (int k = 0; k < 2; ++k) {
        const int e = tok_e[2 * t + k];
        slot_token[offsets[e] + tok_rank[2 * t + k]] = t;
    }
}

// --------------------------------------------------------- shared helpers ---
// Stage a 32(K) x 128(N) fp32 weight tile into Bsl[n][k] (bf16, transposed).
// Thread mapping: 2 adjacent K rows x 8 N cols -> packed dword LDS stores
// (compiles to v_cvt_pk_bf16_f32 + ds_store_2addr_b32).
__device__ __forceinline__ void stage_B(
    bf16 (*Bsl)[LDK], const float* __restrict__ Wbase, int ldw,
    int k0, int n0, int tid)
{
    const int kk = (tid & 15) * 2;        // 0,2,..,30
    const int nb = (tid >> 4) * 8;        // 0,8,..,120
    const float* r0 = Wbase + (size_t)(k0 + kk) * ldw + n0 + nb;
    const float* r1 = r0 + ldw;
    __builtin_prefetch((const char*)r0 + (size_t)BK * ldw * 4, 0, 1);
    const float4 a0 = ((const float4*)r0)[0], a1 = ((const float4*)r0)[1];
    const float4 b0 = ((const float4*)r1)[0], b1 = ((const float4*)r1)[1];
    float lo[8] = {a0.x, a0.y, a0.z, a0.w, a1.x, a1.y, a1.z, a1.w};
    float hi[8] = {b0.x, b0.y, b0.z, b0.w, b1.x, b1.y, b1.z, b1.w};
    #pragma unroll
    for (int j = 0; j < 8; ++j) {
        Pack4B p;
        p.h[0] = (bf16)lo[j];             // (k, n+j)
        p.h[1] = (bf16)hi[j];             // (k+1, n+j)
        *(unsigned*)&Bsl[nb + j][kk] = p.u;   // ds_store_b32 (4B aligned)
    }
}

// Stage one gathered fp32 x-row chunk as bf16 into Asl (one b128 LDS store).
__device__ __forceinline__ void stage_A_cvt(
    bf16 (*Asl)[LDK], const float* __restrict__ xrow, bool rowOk,
    int k0, int sAr, int sAk)
{
    Pack16B p;
    if (rowOk) {
        const float4* xp = (const float4*)(xrow + k0 + sAk);
        const float4 f0 = xp[0], f1 = xp[1];
        p.h[0] = (bf16)f0.x; p.h[1] = (bf16)f0.y;
        p.h[2] = (bf16)f0.z; p.h[3] = (bf16)f0.w;
        p.h[4] = (bf16)f1.x; p.h[5] = (bf16)f1.y;
        p.h[6] = (bf16)f1.z; p.h[7] = (bf16)f1.w;
    } else {
        p.q = make_uint4(0u, 0u, 0u, 0u);
    }
    *(uint4*)&Asl[sAr][sAk] = p.q;
}

// One K-step of wave-level WMMA work: 10x ds_load_b128 pinned above 4 WMMAs
// so the dscnt waits stagger and LDS latency overlaps matrix math.
__device__ __forceinline__ void tile_mma(
    const bf16 (*Asl)[LDK], const bf16 (*Bsl)[LDK], v8f acc[4],
    int wr, int wc, int cN, int kH)
{
    Frag32B a;   // A fragment per §7.12.2 (16x32 bf16)
    {
        const int row = wr * 16 + cN;
        a.q[0] = *(const uint4*)&Asl[row][kH * 8];
        a.q[1] = *(const uint4*)&Asl[row][kH * 8 + 16];
    }
    Frag32B b[4];
    #pragma unroll
    for (int nt = 0; nt < 4; ++nt) {
        const int col = wc * 64 + nt * 16 + cN;
        b[nt].q[0] = *(const uint4*)&Bsl[col][kH * 8];
        b[nt].q[1] = *(const uint4*)&Bsl[col][kH * 8 + 16];
    }
    __builtin_amdgcn_sched_barrier(0);   // keep all loads above the WMMAs
    #pragma unroll
    for (int nt = 0; nt < 4; ++nt)
        acc[nt] = __builtin_amdgcn_wmma_f32_16x16x32_bf16(
            false, a.v, false, b[nt].v, (short)0, acc[nt], false, false);
}

// ------------------------------------------------------------- FFN GEMM 1 ---
// h[slot, :H] = gelu(x[token(slot), :D] @ W1[e] + b1[e]) in bf16
__global__ __launch_bounds__(256) void moe_ffn1(
    const float* __restrict__ x, const float* __restrict__ W1,
    const float* __restrict__ b1, const int* __restrict__ counts,
    const int* __restrict__ offsets, const int* __restrict__ slot_token,
    bf16* __restrict__ h)
{
    __shared__ bf16 Asl[2][BM][LDK];   // ping-pong A tiles
    __shared__ bf16 Bsl[2][BN][LDK];   // ping-pong B tiles ([n][k] layout)

    const int e    = blockIdx.y;
    const int n0   = blockIdx.x * BN;
    const int tid  = threadIdx.x;
    const int lane = tid & 31;
    const int wave = tid >> 5;
    const int wr   = wave & 3;       // wave row  -> 16 M-rows
    const int wc   = wave >> 2;      // wave col  -> 64 N-cols
    const int cN   = lane & 15;
    const int kH   = lane >> 4;      // lane half (K/M split per ISA layout)

    const int cnt   = counts[e];
    const int sbase = offsets[e];
    const float* W1e = W1 + (size_t)e * DDIM * HDIM;

    const int sAr = tid >> 2;          // 0..63 : A row
    const int sAk = (tid & 3) * 8;     // 0,8,16,24 : A k-chunk

    for (int m0 = 0; m0 < cnt; m0 += BM) {
        const bool rowOk = (m0 + sAr) < cnt;
        const int  myTok = rowOk ? slot_token[sbase + m0 + sAr] : 0;
        const float* xrow = x + (size_t)myTok * DDIM;

        v8f acc[4] = {};

        // prologue: stage K-step 0 into buffer 0
        stage_A_cvt(Asl[0], xrow, rowOk, 0, sAr, sAk);
        stage_B(Bsl[0], W1e, HDIM, 0, n0, tid);

        for (int k0 = 0; k0 < DDIM; k0 += BK) {
            const int cur = (k0 / BK) & 1;
            __syncthreads();           // staged tile 'cur' now visible
            if (k0 + BK < DDIM) {      // stage next tile while computing
                stage_A_cvt(Asl[1 - cur], xrow, rowOk, k0 + BK, sAr, sAk);
                stage_B(Bsl[1 - cur], W1e, HDIM, k0 + BK, n0, tid);
            }
            tile_mma(Asl[cur], Bsl[cur], acc, wr, wc, cN, kH);
        }

        // ---- epilogue: bias + exact GELU -> bf16 h
        #pragma unroll
        for (int nt = 0; nt < 4; ++nt) {
            const int col  = n0 + wc * 64 + nt * 16 + cN;
            const float bs = b1[e * HDIM + col];
            #pragma unroll
            for (int i = 0; i < 8; ++i) {
                const int mm = wr * 16 + kH * 8 + i;
                if (m0 + mm < cnt) {
                    float v = acc[nt][i] + bs;
                    v = 0.5f * v * (1.0f + erff(v * 0.70710678118654752f));
                    h[(size_t)(sbase + m0 + mm) * HDIM + col] = (bf16)v;
                }
            }
        }
        __syncthreads();               // protect buffers before next prologue
    }
}

// ------------------------------------------------------------- FFN GEMM 2 ---
// pairout[slot, :D] = h[slot, :H] @ W2[e] + b2[e]   (fp32)
__global__ __launch_bounds__(256) void moe_ffn2(
    const bf16* __restrict__ h, const float* __restrict__ W2,
    const float* __restrict__ b2, const int* __restrict__ counts,
    const int* __restrict__ offsets, float* __restrict__ pairout)
{
    __shared__ bf16 Asl[2][BM][LDK];
    __shared__ bf16 Bsl[2][BN][LDK];

    const int e    = blockIdx.y;
    const int n0   = blockIdx.x * BN;
    const int tid  = threadIdx.x;
    const int lane = tid & 31;
    const int wave = tid >> 5;
    const int wr   = wave & 3;
    const int wc   = wave >> 2;
    const int cN   = lane & 15;
    const int kH   = lane >> 4;

    const int cnt   = counts[e];
    const int sbase = offsets[e];
    const float* W2e = W2 + (size_t)e * HDIM * DDIM;

    const int sAr = tid >> 2;
    const int sAk = (tid & 3) * 8;
    const unsigned ldsA0 = (unsigned)(uintptr_t)&Asl[0][sAr][sAk];
    const unsigned ldsA1 = (unsigned)(uintptr_t)&Asl[1][sAr][sAk];

    for (int m0 = 0; m0 < cnt; m0 += BM) {
        // clamp OOR rows to a valid row: loads garbage-free, stores guarded
        const int arow = (m0 + sAr < cnt) ? (m0 + sAr) : (cnt - 1);
        const bf16* agp = h + (size_t)(sbase + arow) * HDIM + sAk;

        v8f acc[4] = {};

        // prologue: async-DMA A tile 0 (bf16, no VGPR hop) + stage B tile 0
        {
            const unsigned long long ga = (unsigned long long)(uintptr_t)agp;
            asm volatile("global_load_async_to_lds_b128 %0, %1, off"
                         :: "v"(ldsA0), "v"(ga) : "memory");
        }
        stage_B(Bsl[0], W2e, DDIM, 0, n0, tid);

        for (int k0 = 0; k0 < HDIM; k0 += BK) {
            const int cur = (k0 / BK) & 1;
            // drain this wave's async copy for tile 'cur', then barrier
            asm volatile("s_wait_asynccnt 0x0" ::: "memory");
            __syncthreads();
            if (k0 + BK < HDIM) {      // stage next tile while computing
                const unsigned long long ga =
                    (unsigned long long)(uintptr_t)(agp + k0 + BK);
                asm volatile("global_load_async_to_lds_b128 %0, %1, off"
                             :: "v"(cur ? ldsA0 : ldsA1), "v"(ga) : "memory");
                stage_B(Bsl[1 - cur], W2e, DDIM, k0 + BK, n0, tid);
            }
            tile_mma(Asl[cur], Bsl[cur], acc, wr, wc, cN, kH);
        }

        #pragma unroll
        for (int nt = 0; nt < 4; ++nt) {
            const int col  = n0 + wc * 64 + nt * 16 + cN;
            const float bs = b2[e * DDIM + col];
            #pragma unroll
            for (int i = 0; i < 8; ++i) {
                const int mm = wr * 16 + kH * 8 + i;
                if (m0 + mm < cnt)
                    pairout[(size_t)(sbase + m0 + mm) * DDIM + col] = acc[nt][i] + bs;
            }
        }
        __syncthreads();
    }
}

// ---------------------------------------------------------------- combine ---
__global__ __launch_bounds__(256) void moe_combine(
    const float* __restrict__ pairout, const int* __restrict__ tok_e,
    const int* __restrict__ tok_rank, const float* __restrict__ tok_gate,
    const int* __restrict__ offsets, float* __restrict__ y)
{
    const int idx = blockIdx.x * 256 + threadIdx.x;   // over TOKENS * (D/4)
    const int t = idx / (DDIM / 4);
    const int c = idx % (DDIM / 4);
    if (t >= TOKENS) return;
    const int e0 = tok_e[2 * t], e1 = tok_e[2 * t + 1];
    const int s0 = offsets[e0] + tok_rank[2 * t];
    const int s1 = offsets[e1] + tok_rank[2 * t + 1];
    const float g0 = tok_gate[2 * t], g1 = tok_gate[2 * t + 1];
    const float4 a = ((const float4*)(pairout + (size_t)s0 * DDIM))[c];
    const float4 b = ((const float4*)(pairout + (size_t)s1 * DDIM))[c];
    float4 r;
    r.x = g0 * a.x + g1 * b.x;  r.y = g0 * a.y + g1 * b.y;
    r.z = g0 * a.z + g1 * b.z;  r.w = g0 * a.w + g1 * b.w;
    ((float4*)(y + (size_t)t * DDIM))[c] = r;
}

// ----------------------------------------------------------------- launch ---
extern "C" void kernel_launch(void* const* d_in, const int* in_sizes, int n_in,
                              void* d_out, int out_size, void* d_ws, size_t ws_size,
                              hipStream_t stream) {
    (void)in_sizes; (void)n_in; (void)out_size; (void)ws_size;
    const float* x  = (const float*)d_in[0];
    const float* Wg = (const float*)d_in[1];
    const float* bg = (const float*)d_in[2];
    const float* W1 = (const float*)d_in[3];
    const float* b1 = (const float*)d_in[4];
    const float* W2 = (const float*)d_in[5];
    const float* b2 = (const float*)d_in[6];
    float* y = (float*)d_out;

    // workspace carve-up (all 256B-aligned)
    char* ws = (char*)d_ws;
    int*   counts     = (int*)ws;                         ws += 256;
    int*   offsets    = (int*)ws;                         ws += 256;
    int*   tok_e      = (int*)ws;                         ws += (size_t)TOKENS * 2 * 4;
    int*   tok_rank   = (int*)ws;                         ws += (size_t)TOKENS * 2 * 4;
    float* tok_gate   = (float*)ws;                       ws += (size_t)TOKENS * 2 * 4;
    int*   slot_token = (int*)ws;                         ws += (size_t)NPAIR * 4;
    bf16*  hbuf       = (bf16*)ws;                        ws += (size_t)NPAIR * HDIM * 2;
    float* pairout    = (float*)ws;

    moe_zero_counts<<<1, 256, 0, stream>>>(counts);
    moe_routing<<<TOKENS / 8, 256, 0, stream>>>(x, Wg, bg, counts,
                                                tok_e, tok_rank, tok_gate);
    moe_scan<<<1, 32, 0, stream>>>(counts, offsets);
    moe_scatter<<<TOKENS / 256, 256, 0, stream>>>(tok_e, tok_rank, offsets,
                                                  slot_token);
    moe_ffn1<<<dim3(HDIM / BN, NEXP), 256, 0, stream>>>(x, W1, b1, counts,
                                                        offsets, slot_token, hbuf);
    moe_ffn2<<<dim3(DDIM / BN, NEXP), 256, 0, stream>>>(hbuf, W2, b2, counts,
                                                        offsets, pairout);
    moe_combine<<<(TOKENS * (DDIM / 4)) / 256, 256, 0, stream>>>(
        pairout, tok_e, tok_rank, tok_gate, offsets, y);
}
